// Encoder_23845658428153
// MI455X (gfx1250) — compile-verified, matrix operand order checked by reading, not run on hardware
//
#include <hip/hip_runtime.h>

// ---------------------------------------------------------------------------
// CDNA5 (gfx1250) Mamba-FFT encoder.
// All dense projections + DFT matmuls on v_wmma_f32_16x16x32_f16.
// B-tiles staged into LDS with global_load_async_to_lds_b128 (ASYNCcnt).
// All GEMM shapes padded so the hot loop has zero bounds checks.
// ---------------------------------------------------------------------------

typedef _Float16 half_t;
typedef _Float16     v16h __attribute__((ext_vector_type(16)));
typedef float        v8f  __attribute__((ext_vector_type(8)));
typedef unsigned int v4u  __attribute__((ext_vector_type(4)));
typedef unsigned int v2u  __attribute__((ext_vector_type(2)));

#define BM 32
#define BN 64
#define BK 32
#define LDP 8   // LDS row padding (halfs) -> stride 40 halfs = 80B (16B aligned)

__device__ __forceinline__ unsigned lds_off_u32(const void* p)
{
    // generic -> LDS(AS3) addrspacecast, then ptrtoint => byte offset in LDS
    return (unsigned)(unsigned long long)(__attribute__((address_space(3))) const void*)p;
}

// ---------------------------------------------------------------------------
// WMMA GEMM: C[M,N] (f32) = A[M,K](f16,lda) x B (f16; TRANSB ? [N,K] : [K,N])
//            (+C if ACCUM). REQUIRES: M%32==0, N%64==0, K%32==0,
//            lda%4==0, ldb%8==0 (transB) / ldb%8==0 (non-transB).
// blockDim 256 (8 waves); wave tile 16x16; block tile 32x64; K-step 32.
// ---------------------------------------------------------------------------
template <bool TRANSB, bool ACCUM>
__global__ void __launch_bounds__(256)
k_gemm_wmma(const half_t* __restrict__ A, const half_t* __restrict__ B,
            float* __restrict__ C,
            int M, int N, int K, int lda, int ldb, int ldc,
            long long strideA, long long strideB, long long strideC)
{
    __shared__ __align__(16) half_t sA [BM][BK + LDP];  // [m][k]
    __shared__ __align__(16) half_t sBT[BN][BK + LDP];  // [n][k]

    const long long bz = blockIdx.z;
    A += bz * strideA;
    B += bz * strideB;
    C += bz * strideC;

    const int m0   = blockIdx.y * BM;
    const int n0   = blockIdx.x * BN;
    const int tid  = threadIdx.x;
    const int lane = tid & 31;
    const int wave = tid >> 5;           // 0..7
    const int wm   = (wave >> 2) * 16;   // 0 / 16
    const int wn   = (wave & 3) * 16;    // 0..48
    const int lm   = lane & 15;
    const int lh   = lane >> 4;          // half-wave

    // loader decomposition (straight-line, no guards)
    const int am = tid >> 3;             // A row     0..31
    const int ak = (tid & 7) * 4;        // A col x4  0..28
    const int bn = tid >> 2;             // B row     0..63   (TRANSB)
    const int bk = (tid & 3) * 8;        // B col x8  0..24   (TRANSB)
    const int ck = tid >> 3;             // B k-row   0..31   (!TRANSB)
    const int cn = (tid & 7) * 8;        // B n   x8  0..56   (!TRANSB)

    const half_t* pA = &A[(long long)(m0 + am) * lda + ak];
    const half_t* pB = TRANSB ? &B[(long long)(n0 + bn) * ldb + bk]
                              : &B[(long long)ck * ldb + n0 + cn];
    const unsigned sb_off = lds_off_u32(&sBT[bn][bk]);

    v8f acc = {};

    for (int kk = 0; kk < K; kk += BK) {
        // ---- A tile: 8 bytes per thread, vectorized ----
        *(v2u*)&sA[am][ak] = *(const v2u*)(pA + kk);

        if (TRANSB) {
            // ---- B tile: 16B per thread, async global -> LDS (no VGPRs) ----
            unsigned long long ga = (unsigned long long)(const void*)(pB + kk);
            asm volatile("global_load_async_to_lds_b128 %0, %1, off"
                         :: "v"(sb_off), "v"(ga) : "memory");
        } else {
            // ---- B tile: load 8 halfs along n, scatter-transpose to sBT ----
            v4u bv = *(const v4u*)(pB + (long long)kk * ldb);
            const half_t* hv = (const half_t*)&bv;
#pragma unroll
            for (int j = 0; j < 8; ++j) sBT[cn + j][ck] = hv[j];
        }

        // prefetch next K tiles while this one lands
        if (kk + BK < K) {
            __builtin_prefetch((const void*)(pA + kk + BK), 0, 3);
            if (TRANSB) __builtin_prefetch((const void*)(pB + kk + BK), 0, 3);
        }

        if (TRANSB) {
            asm volatile("s_wait_asynccnt 0" ::: "memory");
        }
        __syncthreads();

        // ---- fragments: 2x ds_load_b128 each (CDNA5 16-bit A/B layout) ----
        union FragH { v16h h; v4u q[2]; };
        FragH fa, fb;
        fa.q[0] = *(const v4u*)&sA [wm + lm][lh * 8];
        fa.q[1] = *(const v4u*)&sA [wm + lm][16 + lh * 8];
        fb.q[0] = *(const v4u*)&sBT[wn + lm][lh * 8];
        fb.q[1] = *(const v4u*)&sBT[wn + lm][16 + lh * 8];

        acc = __builtin_amdgcn_wmma_f32_16x16x32_f16(
            false, fa.h, false, fb.h, (short)0, acc, false, false);

        __syncthreads();
    }

    // ---- epilogue: VGPR r -> row m0+wm+r+8*lh, col n0+wn+lm ----
#pragma unroll
    for (int r = 0; r < 8; ++r) {
        long long idx = (long long)(m0 + wm + r + 8 * lh) * ldc + (n0 + wn + lm);
        float v = acc[r];
        if (ACCUM) v += C[idx];
        C[idx] = v;
    }
}

// ---------------------------------------------------------------------------
// DFT matrices (f16), rfft bins padded 129 -> KP=160.
// Fc/Fs: [160,256] (rows 129..159 zero). Ir/Ii: [256,160] (cols >128 zero).
// ---------------------------------------------------------------------------
#define KP 160

__global__ void k_init_dft(half_t* __restrict__ Fc, half_t* __restrict__ Fs,
                           half_t* __restrict__ Ir, half_t* __restrict__ Ii)
{
    const float TWO_PI = 6.283185307179586f;
    int idx = blockIdx.x * blockDim.x + threadIdx.x;
    if (idx < KP * 256) {
        int k = idx / 256, l = idx % 256;
        float c = 0.f, s = 0.f;
        if (k <= 128) {
            float th = TWO_PI * (float)(k * l) / 256.f;
            c = __cosf(th);
            s = -__sinf(th);
        }
        Fc[idx] = (half_t)c;
        Fs[idx] = (half_t)s;
    }
    if (idx < 256 * KP) {
        int l = idx / KP, k = idx % KP;
        float c = 0.f, s = 0.f;
        if (k <= 128) {
            float a  = (k == 0 || k == 128) ? 1.f : 2.f;
            float th = TWO_PI * (float)(k * l) / 256.f;
            c = a * __cosf(th) / 256.f;
            s = -a * __sinf(th) / 256.f;
        }
        Ir[idx] = (half_t)c;
        Ii[idx] = (half_t)s;
    }
}

// ---------------------------------------------------------------------------
// Elementwise helpers
// ---------------------------------------------------------------------------
__global__ void k_f32_to_f16(const float* __restrict__ src,
                             half_t* __restrict__ dst, long long n)
{
    long long i = (long long)blockIdx.x * blockDim.x + threadIdx.x;
    if (i < n) dst[i] = (half_t)src[i];
}

// pad rows: dst[r,c] = (r < rows_src) ? src[r,c] : 0   (both ld == cols)
__global__ void k_pad_rows_f16(const float* __restrict__ src,
                               half_t* __restrict__ dst,
                               int rows_src, int rows_dst, int cols)
{
    long long i = (long long)blockIdx.x * blockDim.x + threadIdx.x;
    if (i >= (long long)rows_dst * cols) return;
    int r = (int)(i / cols);
    dst[i] = (r < rows_src) ? (half_t)src[i] : (half_t)0.f;
}

// pad cols: dst[r, 0..cols_dst) = (c < cols_src) ? src[r*cols_src+c] : 0
__global__ void k_pad_cols_f16(const float* __restrict__ src,
                               half_t* __restrict__ dst,
                               long long rows, int cols_src, int cols_dst)
{
    long long i = (long long)blockIdx.x * blockDim.x + threadIdx.x;
    if (i >= rows * cols_dst) return;
    long long r = i / cols_dst;
    int c = (int)(i % cols_dst);
    dst[i] = (c < cols_src) ? (half_t)src[r * cols_src + c] : (half_t)0.f;
}

// dt slice: dst[r, 0..32) = (c < dtrank) ? dbc[r*64+c] : 0
__global__ void k_slice_pad_f16(const float* __restrict__ src,
                                half_t* __restrict__ dst,
                                long long rows, int dtrank)
{
    long long i = (long long)blockIdx.x * blockDim.x + threadIdx.x;
    if (i >= rows * 32) return;
    long long r = i / 32;
    int c = (int)(i % 32);
    dst[i] = (c < dtrank) ? (half_t)src[r * 64 + c] : (half_t)0.f;
}

// Frequency-domain mask + complex channel weight -> f16 spectra.
__global__ void k_freq_mask(const float* __restrict__ Xr,
                            const float* __restrict__ Xi,
                            const float* __restrict__ cw,
                            const float* __restrict__ lo,
                            const float* __restrict__ hi,
                            half_t* __restrict__ Pr, half_t* __restrict__ Pi,
                            int D, long long total)
{
    long long i = (long long)blockIdx.x * blockDim.x + threadIdx.x;
    if (i >= total) return;
    int d = (int)(i % D);
    int k = (int)((i / D) % KP);
    float lth = *lo, hth = *hi;
    float xr = Xr[i], xi = Xi[i];
    float amp = sqrtf(xr * xr + xi * xi);
    float m = (k <= 128 && amp > lth && amp < hth) ? 1.f : 0.f;
    float wr = cw[2 * d], wi = cw[2 * d + 1];
    Pr[i] = (half_t)(m * (xr * wr - xi * wi));
    Pi[i] = (half_t)(m * (xr * wi + xi * wr));
}

// xf += x; emit f16 forward + L-flipped copies.
__global__ void k_finish_freq(float* __restrict__ xf, const float* __restrict__ x,
                              half_t* __restrict__ x16, half_t* __restrict__ xr16,
                              int L, int D, long long total)
{
    long long i = (long long)blockIdx.x * blockDim.x + threadIdx.x;
    if (i >= total) return;
    int d = (int)(i % D);
    int l = (int)((i / D) % L);
    long long b = i / ((long long)L * D);
    float v = xf[i] + x[i];
    xf[i]  = v;
    x16[i] = (half_t)v;
    xr16[(b * L + (L - 1 - l)) * D + d] = (half_t)v;
}

__device__ __forceinline__ float silu_f(float v)
{
    return v * (1.f / (1.f + __expf(-v)));
}

// Causal depthwise conv (K=4, left pad 3) + SiLU on the x-half of xz.
__global__ void k_dwconv_silu(const float* __restrict__ xz,
                              const float* __restrict__ cw,
                              const float* __restrict__ cb,
                              float* __restrict__ xi, half_t* __restrict__ xi16,
                              int L, int di, long long total)
{
    long long i = (long long)blockIdx.x * blockDim.x + threadIdx.x;
    if (i >= total) return;
    int c = (int)(i % di);
    int l = (int)((i / di) % L);
    long long b = i / ((long long)L * di);
    const int ld = 2 * di;
    float s = cb[c];
#pragma unroll
    for (int j = 0; j < 4; ++j) {
        int ls = l - 3 + j;
        if (ls >= 0) s += cw[c * 4 + j] * xz[(b * L + ls) * (long long)ld + c];
    }
    float v = silu_f(s);
    xi[i]   = v;
    xi16[i] = (half_t)v;
}

// dt = softplus(dt_raw + bias), in place
__global__ void k_softplus_bias(float* __restrict__ dt,
                                const float* __restrict__ bias,
                                int di, long long total)
{
    long long i = (long long)blockIdx.x * blockDim.x + threadIdx.x;
    if (i >= total) return;
    int c = (int)(i % di);
    float v = dt[i] + bias[c];
    dt[i] = (v > 20.f) ? v : log1pf(__expf(v));
}

// Selective scan: one thread per (batch, channel); 16 SSM states in registers.
__global__ void k_scan(const float* __restrict__ dt, const float* __restrict__ xi,
                       const float* __restrict__ dbc, const float* __restrict__ xz,
                       const float* __restrict__ A_log, const float* __restrict__ Dp,
                       float* __restrict__ y, half_t* __restrict__ y16,
                       int B, int L, int di, int dtrank)
{
    int t = blockIdx.x * blockDim.x + threadIdx.x;
    if (t >= B * di) return;
    int b = t / di, c = t % di;

    float a[16], h[16];
#pragma unroll
    for (int s = 0; s < 16; ++s) {
        a[s] = -__expf(A_log[c * 16 + s]);
        h[s] = 0.f;
    }
    float Dc = Dp[c];

    for (int l = 0; l < L; ++l) {
        long long row = (long long)b * L + l;
        float dtv = dt[row * di + c];
        float xiv = xi[row * di + c];
        float zv  = xz[row * (long long)(2 * di) + di + c];
        const float* bc = dbc + row * 64 + dtrank;
        float accv = 0.f;
#pragma unroll
        for (int s = 0; s < 16; ++s) {
            float dA = __expf(dtv * a[s]);
            h[s] = dA * h[s] + dtv * bc[s] * xiv;
            accv += h[s] * bc[16 + s];
        }
        float yv = (accv + Dc * xiv) * silu_f(zv);
        y[row * di + c]   = yv;
        y16[row * di + c] = (half_t)yv;
    }
}

// y0 = LN(y_fwd + flip(y_bwd) + dwconv3(xf) + x). One block per (b,i) row.
__global__ void __launch_bounds__(256)
k_combine1(const float* __restrict__ yf, const float* __restrict__ yb,
           const float* __restrict__ xf, const float* __restrict__ x,
           const float* __restrict__ lw, const float* __restrict__ lb,
           const float* __restrict__ g,  const float* __restrict__ be,
           float* __restrict__ y0, int L, int D)
{
    int row = blockIdx.x;
    int b = row / L, i = row % L;
    int tid = threadIdx.x;
    const float* xr = xf + ((long long)b * L + i) * D;
    float w0 = lw[i * 3], w1 = lw[i * 3 + 1], w2 = lw[i * 3 + 2], bb = lb[i];

    float vals[2], sum = 0.f, sq = 0.f;
#pragma unroll
    for (int e = 0; e < 2; ++e) {
        int m = tid + e * 256;
        float conv = bb + w1 * xr[m];
        if (m > 0)     conv += w0 * xr[m - 1];
        if (m < D - 1) conv += w2 * xr[m + 1];
        float v = yf[((long long)b * L + i) * D + m]
                + yb[((long long)b * L + (L - 1 - i)) * D + m]
                + conv
                + x[((long long)b * L + i) * D + m];
        vals[e] = v;
        sum += v;
        sq  += v * v;
    }
    __shared__ float s1[256], s2[256];
    s1[tid] = sum; s2[tid] = sq;
    __syncthreads();
    for (int off = 128; off > 0; off >>= 1) {
        if (tid < off) { s1[tid] += s1[tid + off]; s2[tid] += s2[tid + off]; }
        __syncthreads();
    }
    float mean = s1[0] / (float)D;
    float var  = s2[0] / (float)D - mean * mean;
    float rstd = rsqrtf(var + 1e-5f);
#pragma unroll
    for (int e = 0; e < 2; ++e) {
        int m = tid + e * 256;
        y0[((long long)b * L + i) * D + m] = (vals[e] - mean) * rstd * g[m] + be[m];
    }
}

// transpose y0 [B,L,D] -> f16 [B,D,L] forward + flipped (along new L=D axis)
__global__ void k_transpose2(const float* __restrict__ y0,
                             half_t* __restrict__ tf, half_t* __restrict__ tr,
                             int L, int D, long long total)
{
    long long i = (long long)blockIdx.x * blockDim.x + threadIdx.x;
    if (i >= total) return;
    int m = (int)(i % D);
    int l = (int)((i / D) % L);
    long long b = i / ((long long)L * D);
    float v = y0[i];
    tf[(b * D + m) * L + l]           = (half_t)v;
    tr[(b * D + (D - 1 - m)) * L + l] = (half_t)v;
}

// out = LN2( (m2f + flip(m2b))^T * y0 + x ). One block per (b,i) row.
__global__ void __launch_bounds__(256)
k_combine2(const float* __restrict__ f2, const float* __restrict__ b2,
           const float* __restrict__ y0, const float* __restrict__ x,
           const float* __restrict__ g,  const float* __restrict__ be,
           float* __restrict__ out, int Din, int Lm)
{
    int row = blockIdx.x;
    int b = row / Din, i = row % Din;
    int tid = threadIdx.x;

    float vals[2], sum = 0.f, sq = 0.f;
#pragma unroll
    for (int e = 0; e < 2; ++e) {
        int m = tid + e * 256;
        float v1 = f2[((long long)b * Lm + m) * Din + i]
                 + b2[((long long)b * Lm + (Lm - 1 - m)) * Din + i];
        float v = v1 * y0[((long long)b * Din + i) * Lm + m]
                + x[((long long)b * Din + i) * Lm + m];
        vals[e] = v;
        sum += v;
        sq  += v * v;
    }
    __shared__ float s1[256], s2[256];
    s1[tid] = sum; s2[tid] = sq;
    __syncthreads();
    for (int off = 128; off > 0; off >>= 1) {
        if (tid < off) { s1[tid] += s1[tid + off]; s2[tid] += s2[tid + off]; }
        __syncthreads();
    }
    float mean = s1[0] / (float)Lm;
    float var  = s2[0] / (float)Lm - mean * mean;
    float rstd = rsqrtf(var + 1e-5f);
#pragma unroll
    for (int e = 0; e < 2; ++e) {
        int m = tid + e * 256;
        out[((long long)b * Din + i) * Lm + m] = (vals[e] - mean) * rstd * g[m] + be[m];
    }
}

// ---------------------------------------------------------------------------
// Host side
// ---------------------------------------------------------------------------
static inline void* ws_take(char*& p, size_t bytes)
{
    void* r = (void*)p;
    p += (bytes + 255) & ~(size_t)255;
    return r;
}

static inline void launch_f32_to_f16(const float* s, half_t* d, long long n,
                                     hipStream_t st)
{
    k_f32_to_f16<<<(int)((n + 255) / 256), 256, 0, st>>>(s, d, n);
}

struct MambaW {
    const float *in_w, *conv_w, *conv_b, *xproj_w, *dt_w, *dt_b, *A_log, *D, *out_w;
};
struct MambaW16 {      // f16 weights; xproj padded to 64 rows, dt_w to 32 cols
    half_t *in_w, *xproj_w, *dt_w, *out_w;
};
struct MambaScratch {
    float* xz;       // [rows, 2*di]
    float* xi;       // [rows, di]
    half_t* xi16;
    float* dbc;      // [rows, 64]
    half_t* dtin16;  // [rows, 32]
    float* dt;       // [rows, di]
    float* y;        // [rows, di]
    half_t* y16;
};

static void run_mamba(hipStream_t st, int B, int L, int d, int di, int dtrank,
                      const half_t* xin16, const MambaW& w, const MambaW16& w16,
                      const MambaScratch& s, float* out)
{
    const long long rows = (long long)B * L;   // multiple of 32
    dim3 blk(256);

    // xz = xin @ in_w^T  -> [rows, 2*di]
    {
        dim3 grid(2 * di / BN, (int)(rows / BM), 1);
        k_gemm_wmma<true, false><<<grid, blk, 0, st>>>(
            xin16, w16.in_w, s.xz, (int)rows, 2 * di, d, d, d, 2 * di, 0, 0, 0);
    }
    // causal depthwise conv + silu
    {
        long long total = rows * di;
        k_dwconv_silu<<<(int)((total + 255) / 256), 256, 0, st>>>(
            s.xz, w.conv_w, w.conv_b, s.xi, s.xi16, L, di, total);
    }
    // dbc = xi @ xproj^T -> [rows, 64]  (xproj zero-padded to 64 rows)
    {
        dim3 grid(1, (int)(rows / BM), 1);
        k_gemm_wmma<true, false><<<grid, blk, 0, st>>>(
            s.xi16, w16.xproj_w, s.dbc, (int)rows, 64, di, di, di, 64, 0, 0, 0);
    }
    // dt input slice -> f16 [rows,32] (zero-padded)
    {
        long long total = rows * 32;
        k_slice_pad_f16<<<(int)((total + 255) / 256), 256, 0, st>>>(
            s.dbc, s.dtin16, rows, dtrank);
    }
    // dt_raw = dtin @ dt_w^T -> [rows, di]   (K padded to 32)
    {
        dim3 grid(di / BN, (int)(rows / BM), 1);
        k_gemm_wmma<true, false><<<grid, blk, 0, st>>>(
            s.dtin16, w16.dt_w, s.dt, (int)rows, di, 32, 32, 32, di, 0, 0, 0);
    }
    // dt = softplus(dt_raw + dt_b)
    {
        long long total = rows * di;
        k_softplus_bias<<<(int)((total + 255) / 256), 256, 0, st>>>(
            s.dt, w.dt_b, di, total);
    }
    // selective scan + gating
    {
        int total = B * di;
        k_scan<<<(total + 255) / 256, 256, 0, st>>>(
            s.dt, s.xi, s.dbc, s.xz, w.A_log, w.D, s.y, s.y16, B, L, di, dtrank);
    }
    // out = y @ out_w^T -> [rows, d]
    {
        dim3 grid(d / BN, (int)(rows / BM), 1);
        k_gemm_wmma<true, false><<<grid, blk, 0, st>>>(
            s.y16, w16.out_w, out, (int)rows, d, di, di, di, d, 0, 0, 0);
    }
}

static MambaW load_mw(void* const* d_in, int base)
{
    MambaW w;
    w.in_w    = (const float*)d_in[base + 0];
    w.conv_w  = (const float*)d_in[base + 1];
    w.conv_b  = (const float*)d_in[base + 2];
    w.xproj_w = (const float*)d_in[base + 3];
    w.dt_w    = (const float*)d_in[base + 4];
    w.dt_b    = (const float*)d_in[base + 5];
    w.A_log   = (const float*)d_in[base + 6];
    w.D       = (const float*)d_in[base + 7];
    w.out_w   = (const float*)d_in[base + 8];
    return w;
}

// convert one mamba weight set to f16 (with padding for xproj/dt_w)
static void convert_mw(hipStream_t st, const MambaW& w, const MambaW16& w16,
                       int d, int di, int dtrank)
{
    launch_f32_to_f16(w.in_w,  w16.in_w,  (long long)2 * di * d, st);
    {
        long long total = (long long)64 * di;
        k_pad_rows_f16<<<(int)((total + 255) / 256), 256, 0, st>>>(
            w.xproj_w, w16.xproj_w, dtrank + 32, 64, di);
    }
    {
        long long total = (long long)di * 32;
        k_pad_cols_f16<<<(int)((total + 255) / 256), 256, 0, st>>>(
            w.dt_w, w16.dt_w, di, dtrank, 32);
    }
    launch_f32_to_f16(w.out_w, w16.out_w, (long long)d * di, st);
}

extern "C" void kernel_launch(void* const* d_in, const int* in_sizes, int n_in,
                              void* d_out, int out_size, void* d_ws, size_t ws_size,
                              hipStream_t stream)
{
    (void)in_sizes; (void)n_in; (void)out_size; (void)ws_size;

    const int B = 8, ENC = 256, DM = 512;        // x: [B, 256, 512]

    const float* x      = (const float*)d_in[0];
    const float* cw     = (const float*)d_in[1];
    const float* lo_thr = (const float*)d_in[2];
    const float* hi_thr = (const float*)d_in[3];
    const float* lconvw = (const float*)d_in[4];
    const float* lconvb = (const float*)d_in[5];
    const float* ln1w   = (const float*)d_in[6];
    const float* ln1b   = (const float*)d_in[7];
    const float* ln2w   = (const float*)d_in[8];
    const float* ln2b   = (const float*)d_in[9];
    MambaW m1f = load_mw(d_in, 10);
    MambaW m1b = load_mw(d_in, 19);
    MambaW m2f = load_mw(d_in, 28);
    MambaW m2b = load_mw(d_in, 37);
    float* out = (float*)d_out;

    // ---- workspace layout ----
    char* wp = (char*)d_ws;
    half_t* Fc = (half_t*)ws_take(wp, (size_t)KP * 256 * 2);
    half_t* Fs = (half_t*)ws_take(wp, (size_t)KP * 256 * 2);
    half_t* Ir = (half_t*)ws_take(wp, (size_t)256 * KP * 2);
    half_t* Ii = (half_t*)ws_take(wp, (size_t)256 * KP * 2);

    const long long NX = (long long)B * ENC * DM;   // 1M elems
    half_t* x16   = (half_t*)ws_take(wp, NX * 2);
    float*  Xr    = (float*) ws_take(wp, (size_t)B * KP * DM * 4);
    float*  Xi    = (float*) ws_take(wp, (size_t)B * KP * DM * 4);
    half_t* Pr    = (half_t*)ws_take(wp, (size_t)B * KP * DM * 2);
    half_t* Pi    = (half_t*)ws_take(wp, (size_t)B * KP * DM * 2);
    float*  xf    = (float*) ws_take(wp, NX * 4);
    half_t* xf16  = (half_t*)ws_take(wp, NX * 2);
    half_t* xfr16 = (half_t*)ws_take(wp, NX * 2);
    float*  yfwd1 = (float*) ws_take(wp, NX * 4);
    float*  ybwd1 = (float*) ws_take(wp, NX * 4);
    float*  y0    = (float*) ws_take(wp, NX * 4);
    half_t* tfwd  = (half_t*)ws_take(wp, NX * 2);
    half_t* trev  = (half_t*)ws_take(wp, NX * 2);
    float*  yfwd2 = (float*) ws_take(wp, NX * 4);
    float*  ybwd2 = (float*) ws_take(wp, NX * 4);

    // f16 weights (m1: d=512 di=1024 rank=32 ; m2: d=256 di=512 rank=16)
    MambaW16 w1f, w1b, w2f, w2b;
    w1f.in_w    = (half_t*)ws_take(wp, (size_t)2048 * 512 * 2);
    w1f.xproj_w = (half_t*)ws_take(wp, (size_t)64 * 1024 * 2);
    w1f.dt_w    = (half_t*)ws_take(wp, (size_t)1024 * 32 * 2);
    w1f.out_w   = (half_t*)ws_take(wp, (size_t)512 * 1024 * 2);
    w1b.in_w    = (half_t*)ws_take(wp, (size_t)2048 * 512 * 2);
    w1b.xproj_w = (half_t*)ws_take(wp, (size_t)64 * 1024 * 2);
    w1b.dt_w    = (half_t*)ws_take(wp, (size_t)1024 * 32 * 2);
    w1b.out_w   = (half_t*)ws_take(wp, (size_t)512 * 1024 * 2);
    w2f.in_w    = (half_t*)ws_take(wp, (size_t)1024 * 256 * 2);
    w2f.xproj_w = (half_t*)ws_take(wp, (size_t)64 * 512 * 2);
    w2f.dt_w    = (half_t*)ws_take(wp, (size_t)512 * 32 * 2);
    w2f.out_w   = (half_t*)ws_take(wp, (size_t)256 * 512 * 2);
    w2b.in_w    = (half_t*)ws_take(wp, (size_t)1024 * 256 * 2);
    w2b.xproj_w = (half_t*)ws_take(wp, (size_t)64 * 512 * 2);
    w2b.dt_w    = (half_t*)ws_take(wp, (size_t)512 * 32 * 2);
    w2b.out_w   = (half_t*)ws_take(wp, (size_t)256 * 512 * 2);

    // shared mamba scratch (sized for the larger instance)
    MambaScratch ms;
    ms.xz     = (float*) ws_take(wp, (size_t)4194304 * 4);
    ms.xi     = (float*) ws_take(wp, (size_t)2097152 * 4);
    ms.xi16   = (half_t*)ws_take(wp, (size_t)2097152 * 2);
    ms.dbc    = (float*) ws_take(wp, (size_t)4096 * 64 * 4);
    ms.dtin16 = (half_t*)ws_take(wp, (size_t)4096 * 32 * 2);
    ms.dt     = (float*) ws_take(wp, (size_t)2097152 * 4);
    ms.y      = (float*) ws_take(wp, (size_t)2097152 * 4);
    ms.y16    = (half_t*)ws_take(wp, (size_t)2097152 * 2);

    // ---- 0) DFT matrices + f16 conversions ----
    k_init_dft<<<(KP * 256 + 255) / 256, 256, 0, stream>>>(Fc, Fs, Ir, Ii);
    launch_f32_to_f16(x, x16, NX, stream);
    convert_mw(stream, m1f, w1f, 512, 1024, 32);
    convert_mw(stream, m1b, w1b, 512, 1024, 32);
    convert_mw(stream, m2f, w2f, 256, 512, 16);
    convert_mw(stream, m2b, w2b, 256, 512, 16);

    // ---- 1) frequency filter: forward DFT (batched WMMA GEMMs) ----
    {
        dim3 blk(256);
        dim3 grid(DM / BN, KP / BM, B);
        k_gemm_wmma<false, false><<<grid, blk, 0, stream>>>(
            Fc, x16, Xr, KP, DM, 256, 256, DM, DM,
            0LL, (long long)ENC * DM, (long long)KP * DM);
        k_gemm_wmma<false, false><<<grid, blk, 0, stream>>>(
            Fs, x16, Xi, KP, DM, 256, 256, DM, DM,
            0LL, (long long)ENC * DM, (long long)KP * DM);
    }
    // mask + complex weight
    {
        long long total = (long long)B * KP * DM;
        k_freq_mask<<<(int)((total + 255) / 256), 256, 0, stream>>>(
            Xr, Xi, cw, lo_thr, hi_thr, Pr, Pi, DM, total);
    }
    // inverse DFT (accumulating pair) + residual
    {
        dim3 blk(256);
        dim3 grid(DM / BN, ENC / BM, B);
        k_gemm_wmma<false, false><<<grid, blk, 0, stream>>>(
            Ir, Pr, xf, ENC, DM, KP, KP, DM, DM,
            0LL, (long long)KP * DM, (long long)ENC * DM);
        k_gemm_wmma<false, true><<<grid, blk, 0, stream>>>(
            Ii, Pi, xf, ENC, DM, KP, KP, DM, DM,
            0LL, (long long)KP * DM, (long long)ENC * DM);
        k_finish_freq<<<(int)((NX + 255) / 256), 256, 0, stream>>>(
            xf, x, xf16, xfr16, ENC, DM, NX);
    }

    // ---- 2) bidirectional Mamba over L=256, d=512 ----
    run_mamba(stream, B, ENC, 512, 1024, 32, xf16,  m1f, w1f, ms, yfwd1);
    run_mamba(stream, B, ENC, 512, 1024, 32, xfr16, m1b, w1b, ms, ybwd1);

    // ---- 3) combine + LN1 ----
    k_combine1<<<B * ENC, 256, 0, stream>>>(
        yfwd1, ybwd1, xf, x, lconvw, lconvb, ln1w, ln1b, y0, ENC, DM);

    // ---- 4) transpose, bidirectional Mamba over L=512, d=256 ----
    k_transpose2<<<(int)((NX + 255) / 256), 256, 0, stream>>>(
        y0, tfwd, trev, ENC, DM, NX);
    run_mamba(stream, B, DM, 256, 512, 16, tfwd, m2f, w2f, ms, yfwd2);
    run_mamba(stream, B, DM, 256, 512, 16, trev, m2b, w2b, ms, ybwd2);

    // ---- 5) combine + LN2 -> output ----
    k_combine2<<<B * ENC, 256, 0, stream>>>(
        yfwd2, ybwd2, y0, x, ln2w, ln2b, out, ENC, DM);
}
